// MultiHeadSelfAttention_22436909154569
// MI455X (gfx1250) — matrix operand order repaired and checked by reading, not compiled, not run on hardware
//
#include <hip/hip_runtime.h>
#include <math.h>
#include <stdint.h>

typedef __bf16 bf16_t;
typedef __attribute__((ext_vector_type(16))) __bf16 v16bf;
typedef __attribute__((ext_vector_type(8)))  __bf16 v8bf;
typedef __attribute__((ext_vector_type(8)))  float  v8f;

#define BB  2
#define SS  2048
#define DD  1024
#define HH  16
#define DKK 64

// ---- CDNA5 async memory->LDS helpers (ASYNCcnt-tracked DMA, no VGPR data) ----
__device__ __forceinline__ void async_load_b128(unsigned lds_off, const void* gptr) {
    asm volatile("global_load_async_to_lds_b128 %0, %1, off"
                 :: "v"(lds_off), "v"((unsigned long long)(uintptr_t)gptr)
                 : "memory");
}
__device__ __forceinline__ void wait_async_le2() {
    asm volatile("s_wait_asynccnt 0x2" ::: "memory");
}
__device__ __forceinline__ void wait_async_0() {
    asm volatile("s_wait_asynccnt 0x0" ::: "memory");
}

// ---------------------------------------------------------------------------
// Projection: out = relu?(X @ W^T), X:[4096,1024] f32, W:[1024,1024] f32.
// Block = 8 waves sharing one 64-col N-group; W slice staged in LDS as bf16
// (converted once per block instead of once per wave). Each wave computes a
// 16x64 tile with v_wmma_f32_16x16x32_bf16.
// Output bf16, layout [B,H,S,DK] (transpose_out=0) or [B,H,DK,S] (=1).
// ---------------------------------------------------------------------------
__global__ __launch_bounds__(256) void proj_kernel(
    const float* __restrict__ X, const float* __restrict__ W,
    bf16_t* __restrict__ out, int do_relu, int transpose_out)
{
    __shared__ bf16_t sW[64 * 32];   // 4 KB: W[n0..n0+63][kk..kk+31] in bf16

    const int tid  = threadIdx.x;
    const int lane = tid & 31;
    const int wv   = tid >> 5;
    const int nl   = lane & 15;
    const int half = lane >> 4;

    const int mgroup = blockIdx.x >> 4;       // 0..31
    const int n0     = (blockIdx.x & 15) * 64;
    const int mtile  = mgroup * 8 + wv;       // 0..255 (per-wave M tile)

    const int wr = tid >> 2;                  // 0..63 : W row within N slice
    const int wc = (tid & 3) * 8;             // 0,8,16,24

    v8f acc[4] = {{}, {}, {}, {}};
    const float* xrow = X + (size_t)(mtile * 16 + nl) * DD;
    const float* wrow = W + (size_t)(n0 + wr) * DD + wc;

    for (int kk = 0; kk < DD; kk += 32) {
        // ---- cooperative W slice f32 -> bf16 into LDS (8 elems / thread)
        v8f wf = *(const v8f*)(wrow + kk);
        v8bf wb;
#pragma unroll
        for (int i = 0; i < 8; ++i) wb[i] = (bf16_t)wf[i];
        *(v8bf*)(sW + wr * 32 + wc) = wb;
        __syncthreads();

        // ---- A fragment 16x32 from X (per-wave, f32 -> bf16)
        v8f x0 = *(const v8f*)(xrow + kk + 16 * half);
        v8f x1 = *(const v8f*)(xrow + kk + 16 * half + 8);
        v16bf a;
#pragma unroll
        for (int i = 0; i < 8; ++i) { a[i] = (bf16_t)x0[i]; a[8 + i] = (bf16_t)x1[i]; }

        // ---- 4 WMMAs, B fragments straight from LDS
#pragma unroll
        for (int j = 0; j < 4; ++j) {
            v16bf bf = *(const v16bf*)(sW + (j * 16 + nl) * 32 + 16 * half);
            acc[j] = __builtin_amdgcn_wmma_f32_16x16x32_bf16(
                false, a, false, bf, (short)0, acc[j], false, false);
        }
        __syncthreads();   // protect sW before next iteration's overwrite
    }

#pragma unroll
    for (int j = 0; j < 4; ++j)
#pragma unroll
        for (int r = 0; r < 8; ++r) {
            float v = acc[j][r];
            if (do_relu) v = fmaxf(v, 0.0f);
            int row = mtile * 16 + r + 8 * half;    // global row in [0,4096)
            int b = row >> 11;                      // / S
            int s = row & (SS - 1);
            int o = n0 + j * 16 + nl;
            int h  = o >> 6;
            int dk = o & (DKK - 1);
            size_t addr = transpose_out
                ? (((size_t)(b * HH + h) * DKK + dk) * SS + s)
                : (((size_t)(b * HH + h) * SS + s) * DKK + dk);
            out[addr] = (bf16_t)v;
        }
}

// ---------------------------------------------------------------------------
// Flash attention: block = 8 waves, each owning a 16-query tile of one (b,h).
// K/V tiles are double-buffered in LDS and prefetched with
// global_load_async_to_lds_b128 (ASYNCcnt) so tile DMA overlaps compute.
// Per 32-key iteration: 4 WMMA scores + online softmax + P transpose through
// per-wave LDS + 4 WMMA for P.V.
// ---------------------------------------------------------------------------
__global__ __launch_bounds__(256) void attn_kernel(
    const bf16_t* __restrict__ Qh, const bf16_t* __restrict__ Kh,
    const bf16_t* __restrict__ Vt, const int* __restrict__ mask,
    float* __restrict__ out)
{
    __shared__ float  smask[SS];          // additive mask bias, 8 KB
    __shared__ bf16_t sP[8][16 * 32];     // per-wave P staging, 8 KB
    __shared__ bf16_t sK[2][32 * 64];     // K tile [key][feat], 2 x 4 KB
    __shared__ bf16_t sV[2][64 * 32];     // V tile [feat][key], 2 x 4 KB

    const int tid  = threadIdx.x;
    const int lane = tid & 31;
    const int wv   = tid >> 5;
    const int nl   = lane & 15;
    const int half = lane >> 4;
    const int bh   = blockIdx.y;
    const int b    = bh >> 4;             // / H
    const int h    = bh & (HH - 1);
    const int qtile = blockIdx.x * 8 + wv;

    // cooperative-tile decomposition: 16 B per thread per tile
    const int kRow = tid >> 3, kCol = (tid & 7) * 8;   // K: 32 rows x 64 feats
    const int vRow = tid >> 2, vCol = (tid & 3) * 8;   // V: 64 feats x 32 keys
    const bf16_t* kSrc = Kh + ((size_t)bh * SS + kRow) * DKK + kCol;
    const bf16_t* vSrc = Vt + ((size_t)bh * DKK + vRow) * SS + vCol;
    unsigned kDst[2], vDst[2];
#pragma unroll
    for (int p = 0; p < 2; ++p) {
        kDst[p] = (unsigned)(uintptr_t)(&sK[p][0] + kRow * 64 + kCol);
        vDst[p] = (unsigned)(uintptr_t)(&sV[p][0] + vRow * 32 + vCol);
    }

    // prologue: async-prefetch tile 0 into buffer 0
    async_load_b128(kDst[0], kSrc);
    async_load_b128(vDst[0], vSrc);

    for (int i = tid; i < SS; i += 256)
        smask[i] = (mask[b * SS + i] == 0) ? -1.0e30f : 0.0f;

    // Q A-fragments (16 rows x 64 feats -> two 16x32 frags), reused all iterations
    const bf16_t* qrow = Qh + ((size_t)bh * SS + qtile * 16 + nl) * DKK;
    v16bf aq[2];
#pragma unroll
    for (int f = 0; f < 2; ++f) {
        v8bf lo = *(const v8bf*)(qrow + 32 * f + 16 * half);
        v8bf hi = *(const v8bf*)(qrow + 32 * f + 16 * half + 8);
#pragma unroll
        for (int i = 0; i < 8; ++i) { aq[f][i] = lo[i]; aq[f][8 + i] = hi[i]; }
    }

    v8f acc[4] = {{}, {}, {}, {}};
    float m_run[8], l_run[8];
#pragma unroll
    for (int r = 0; r < 8; ++r) { m_run[r] = -1.0e30f; l_run[r] = 0.0f; }

    const float kscale = 0.125f * 1.44269504088896340736f;  // 1/sqrt(DK) * log2(e)
    bf16_t* sPw = &sP[wv][0];

    for (int kb = 0; kb < SS / 32; ++kb) {
        const int kbase = kb * 32;
        const int cur = kb & 1;

        // ---- prefetch next tile into the other buffer, then wait for current.
        // Async loads complete in order: with 2 new in flight, asynccnt<=2
        // guarantees the previous iteration's 2 loads (current tile) are done.
        if (kb < SS / 32 - 1) {
            const int nxt = cur ^ 1;
            async_load_b128(kDst[nxt], kSrc + (size_t)(kbase + 32) * DKK);
            async_load_b128(vDst[nxt], vSrc + (kbase + 32));
            wait_async_le2();
        } else {
            wait_async_0();
        }
        __syncthreads();   // all threads' pieces of the current tile visible

        const bf16_t* sKc = &sK[cur][0];
        const bf16_t* sVc = &sV[cur][0];

        // ---- scores: two 16x16 key blocks, K-dim = 64 feats (2 WMMA each)
        v8f sc[2];
#pragma unroll
        for (int blk = 0; blk < 2; ++blk) {
            const bf16_t* kr = sKc + (blk * 16 + nl) * 64 + 16 * half;
            v16bf b0 = *(const v16bf*)(kr);
            v16bf b1 = *(const v16bf*)(kr + 32);
            v8f c = {};
            c = __builtin_amdgcn_wmma_f32_16x16x32_bf16(false, aq[0], false, b0, (short)0, c, false, false);
            c = __builtin_amdgcn_wmma_f32_16x16x32_bf16(false, aq[1], false, b1, (short)0, c, false, false);
            sc[blk] = c;
        }

        // ---- online softmax (log2 domain, scale folded in)
        const float bias0 = smask[kbase + nl];
        const float bias1 = smask[kbase + 16 + nl];
#pragma unroll
        for (int r = 0; r < 8; ++r) {
            float t0 = sc[0][r] * kscale + bias0;
            float t1 = sc[1][r] * kscale + bias1;
            float mx = fmaxf(t0, t1);
            mx = fmaxf(mx, __shfl_xor(mx, 1, 16));
            mx = fmaxf(mx, __shfl_xor(mx, 2, 16));
            mx = fmaxf(mx, __shfl_xor(mx, 4, 16));
            mx = fmaxf(mx, __shfl_xor(mx, 8, 16));
            float m_new = fmaxf(m_run[r], mx);
            float alpha = exp2f(m_run[r] - m_new);
            m_run[r] = m_new;
            float p0 = exp2f(t0 - m_new);
            float p1 = exp2f(t1 - m_new);
            float sum = p0 + p1;
            sum += __shfl_xor(sum, 1, 16);
            sum += __shfl_xor(sum, 2, 16);
            sum += __shfl_xor(sum, 4, 16);
            sum += __shfl_xor(sum, 8, 16);
            l_run[r] = l_run[r] * alpha + sum;
#pragma unroll
            for (int j = 0; j < 4; ++j) acc[j][r] *= alpha;
            // C-layout -> LDS row-major P[16][32]
            int m = r + 8 * half;
            sPw[m * 32 + nl]      = (bf16_t)p0;
            sPw[m * 32 + 16 + nl] = (bf16_t)p1;
        }

        // ---- reload P as A fragment (same-wave LDS, kept in order by DScnt)
        v16bf ap;
        {
            v8bf lo = *(const v8bf*)(sPw + nl * 32 + 16 * half);
            v8bf hi = *(const v8bf*)(sPw + nl * 32 + 16 * half + 8);
#pragma unroll
            for (int i = 0; i < 8; ++i) { ap[i] = lo[i]; ap[8 + i] = hi[i]; }
        }

        // ---- out += P (16x32) @ V (32x64), V B-fragments from LDS
#pragma unroll
        for (int j = 0; j < 4; ++j) {
            v16bf bv = *(const v16bf*)(sVc + (j * 16 + nl) * 32 + 16 * half);
            acc[j] = __builtin_amdgcn_wmma_f32_16x16x32_bf16(
                false, ap, false, bv, (short)0, acc[j], false, false);
        }
        __syncthreads();   // all waves done reading buf[cur] before it is re-targeted
    }

    // ---- normalize + store: out[b, s, h*64 + feat] (f32)
#pragma unroll
    for (int r = 0; r < 8; ++r) {
        float inv = 1.0f / l_run[r];
        int s = qtile * 16 + r + 8 * half;
        float* orow = out + ((size_t)b * SS + s) * DD + h * DKK;
#pragma unroll
        for (int j = 0; j < 4; ++j)
            orow[j * 16 + nl] = acc[j][r] * inv;
    }
}

// ---------------------------------------------------------------------------
extern "C" void kernel_launch(void* const* d_in, const int* in_sizes, int n_in,
                              void* d_out, int out_size, void* d_ws, size_t ws_size,
                              hipStream_t stream) {
    (void)in_sizes; (void)n_in; (void)out_size; (void)ws_size;

    const float* q    = (const float*)d_in[0];
    const float* k    = (const float*)d_in[1];
    const float* v    = (const float*)d_in[2];
    const int*   mask = (const int*)d_in[3];
    const float* Wq   = (const float*)d_in[4];
    const float* Wk   = (const float*)d_in[5];
    const float* Wv   = (const float*)d_in[6];
    float* out = (float*)d_out;

    const size_t HSZ = (size_t)BB * HH * SS * DKK;   // 4,194,304 elems
    bf16_t* Qh = (bf16_t*)d_ws;
    bf16_t* Kh = Qh + HSZ;
    bf16_t* Vt = Kh + HSZ;                            // total ws use: 24 MB

    dim3 pb(256);
    dim3 pg((BB * SS / 16 / 8) * (DD / 64));          // 32 mgroups * 16 ngroups = 512
    proj_kernel<<<pg, pb, 0, stream>>>(q, Wq, Qh, /*relu=*/1, /*transpose=*/0);
    proj_kernel<<<pg, pb, 0, stream>>>(k, Wk, Kh, /*relu=*/0, /*transpose=*/0);
    proj_kernel<<<pg, pb, 0, stream>>>(v, Wv, Vt, /*relu=*/0, /*transpose=*/1);

    dim3 ag(SS / 16 / 8, BB * HH);                    // (16, 32) blocks of 8 waves
    attn_kernel<<<ag, dim3(256), 0, stream>>>(Qh, Kh, Vt, mask, out);
}
